// MultiHeadAttention_60043642798395
// MI455X (gfx1250) — matrix operand order repaired and checked by reading, not compiled, not run on hardware
//
#include <hip/hip_runtime.h>

// ---------------------------------------------------------------------------
// MI455X (gfx1250) linear-attention pipeline, f16 WMMA with f32 accumulate.
// GEMM tiles move via global_load_async_to_lds_b128 (ASYNCcnt DMA), double
// buffered against the WMMA pipe.
// ---------------------------------------------------------------------------

typedef __attribute__((ext_vector_type(16))) _Float16 v16h;
typedef __attribute__((ext_vector_type(8)))  _Float16 v8h;
typedef __attribute__((ext_vector_type(4)))  _Float16 v4h;
typedef __attribute__((ext_vector_type(8)))  float    v8f;

static constexpr int BB = 4;
static constexpr int TT = 4096;
static constexpr int EE = 2048;
static constexpr int HH = 16;
static constexpr int DD = 128;
static constexpr int BT = BB * TT;       // 16384 rows
static constexpr int SPLIT = 8;          // T-split for the KV reduction

// ---- workspace layout (bytes) ----
static constexpr size_t O_X16   = 0;                         // 64 MB (reused as attn16)
static constexpr size_t O_WQ16  = O_X16  + (size_t)BT * EE * 2;
static constexpr size_t O_WK16  = O_WQ16 + (size_t)EE * EE * 2;
static constexpr size_t O_WV16  = O_WK16 + (size_t)EE * EE * 2;
static constexpr size_t O_WO16  = O_WV16 + (size_t)EE * EE * 2;
static constexpr size_t O_Q16   = O_WO16 + (size_t)EE * EE * 2;
static constexpr size_t O_K16   = O_Q16  + (size_t)BT * EE * 2;
static constexpr size_t O_V16   = O_K16  + (size_t)BT * EE * 2;
static constexpr size_t O_KVP   = O_V16  + (size_t)BT * EE * 2;               // f32 partials
static constexpr size_t O_KVT   = O_KVP  + (size_t)SPLIT * 64 * DD * DD * 4;  // f16 KV^T

// ---------------------------------------------------------------------------
// fp32 -> fp16 conversion (vectorized, exact grids)
// ---------------------------------------------------------------------------
__global__ void cvt_f32_f16(const float* __restrict__ s, _Float16* __restrict__ d, long n) {
    long i = ((long)blockIdx.x * 256 + threadIdx.x) * 4;
    if (i + 3 < n) {
        float4 v = *(const float4*)(s + i);
        v4h o;
        o[0] = (_Float16)v.x; o[1] = (_Float16)v.y;
        o[2] = (_Float16)v.z; o[3] = (_Float16)v.w;
        *(v4h*)(d + i) = o;
    }
}

// ---------------------------------------------------------------------------
// WMMA fragment loads in this file's contiguous k order (k = 16*half + s), same for A and B
//   A (16x32 f16): lane m<16 : V0-7 <- row m, K 0..15
//                  lane m+16: V0-7 <- row m, K 16..31
//   B (32x16 f16): lane n<16 : V0-7 <- col n, K 0..15
//                  lane n+16: V0-7 <- col n, K 16..31
// Tiles are kept in LDS as MxK (A) and NxK (B^T) row-major, so both loads are
// contiguous 16B/32B chunks per lane.
// ---------------------------------------------------------------------------
__device__ __forceinline__ v16h load_a_frag(const _Float16* sA, int row, int pitch,
                                            int kbase, int half) {
    const _Float16* p = sA + row * pitch + kbase + half * 16;
    v8h lo = *(const v8h*)p;
    v8h hi = *(const v8h*)(p + 8);
    return __builtin_shufflevector(lo, hi, 0,1,2,3,4,5,6,7,8,9,10,11,12,13,14,15);
}

__device__ __forceinline__ v16h load_b_frag(const _Float16* sB, int row, int pitch,
                                            int kbase, int half) {
    return *(const v16h*)(sB + row * pitch + kbase + half * 16);
}

// One 32-wide K-step for a wave-owned 32x64 C subtile: 2 A-frags x 4 B-frags,
// 8x v_wmma_f32_16x16x32_f16.
__device__ __forceinline__ void mma_step(const _Float16* sA, int pa, int ka,
                                         const _Float16* sB, int pb, int kb,
                                         int wm, int wn, int lane, v8f acc[2][4]) {
    const int half = lane >> 4, ml = lane & 15;
    v16h af[2], bf[4];
#pragma unroll
    for (int am = 0; am < 2; ++am)
        af[am] = load_a_frag(sA, wm * 32 + am * 16 + ml, pa, ka, half);
#pragma unroll
    for (int bn = 0; bn < 4; ++bn)
        bf[bn] = load_b_frag(sB, wn * 64 + bn * 16 + ml, pb, kb, half);
#pragma unroll
    for (int am = 0; am < 2; ++am)
#pragma unroll
        for (int bn = 0; bn < 4; ++bn)
            acc[am][bn] = __builtin_amdgcn_wmma_f32_16x16x32_f16(
                false, af[am], false, bf[bn], (short)0, acc[am][bn], false, false);
}

// ---------------------------------------------------------------------------
// Async DMA of one 128x32-f16 tile slice into LDS (pitch 40 f16 = 80 B).
// Each wave copies its own 16 rows: 2x global_load_async_to_lds_b128
// (32 lanes x 16 B = 512 B per op; 4 lanes per 64 B row -> 8 rows per op).
// LDS destination address = low 32 bits of the generic pointer.
// ---------------------------------------------------------------------------
__device__ __forceinline__ void async_copy_tile(const _Float16* g, int ld,
                                                int row0, int k0,
                                                unsigned ldsbase, int w, int lane) {
    const int rl = lane >> 2;   // 0..7 : row within the 8-row group
    const int cl = lane & 3;    // 0..3 : 16-byte chunk within the row
#pragma unroll
    for (int i = 0; i < 2; ++i) {
        const int row = w * 16 + i * 8 + rl;
        const unsigned lds = ldsbase + (unsigned)(row * 80 + cl * 16);
        const _Float16* gp = g + (size_t)(row0 + row) * ld + k0 + cl * 8;
        asm volatile("global_load_async_to_lds_b128 %0, %1, off"
                     :: "v"(lds), "v"(gp)
                     : "memory");
    }
}

__device__ __forceinline__ void wait_async4() {
    asm volatile("s_wait_asynccnt 4" ::: "memory");
}
__device__ __forceinline__ void wait_async0() {
    asm volatile("s_wait_asynccnt 0" ::: "memory");
}

// ---------------------------------------------------------------------------
// Y[M,N] = A[M,K] * Bt[N,K]^T + bias    (A row-major, Bt row-major == W layout)
// OUTF32=0: f16 output to outh; OUTF32=1: f32 output to outf.
// Block: 128x128 tile, BK=32, 256 threads (8 waves, 4x2 wave grid),
// double-buffered async-to-LDS tile pipeline.
// LDS arena layout: A0 | B0 | A1 | B1, each 128x40 f16 (10,240 B).
// ---------------------------------------------------------------------------
template <int OUTF32>
__global__ __launch_bounds__(256) void gemm_nt(const _Float16* __restrict__ A,
                                               const _Float16* __restrict__ Bt,
                                               const float* __restrict__ bias,
                                               _Float16* __restrict__ outh,
                                               float* __restrict__ outf,
                                               int K, int lda, int ldb, int ldo) {
    constexpr int TSZ = 128 * 40;                 // elements per tile buffer
    __shared__ _Float16 smem[4 * TSZ];
    const unsigned ldsBase = (unsigned)(size_t)(void*)smem;   // scalar runtime cast

    const int tid = threadIdx.x, lane = tid & 31, w = tid >> 5;
    const int wm = w & 3, wn = w >> 2;
    const int m0 = blockIdx.y * 128, n0 = blockIdx.x * 128;

    v8f acc[2][4] = {};
    const int nk = K / 32;

    // prime buffer 0
    async_copy_tile(A,  lda, m0, 0, ldsBase,         w, lane);
    async_copy_tile(Bt, ldb, n0, 0, ldsBase + 10240u, w, lane);

    for (int kt = 0; kt < nk; ++kt) {
        if (kt + 1 < nk) {   // issue next tile into the other buffer
            const int kn = (kt + 1) * 32;
            const unsigned nbase = ldsBase + (unsigned)(((kt + 1) & 1) * 20480);
            async_copy_tile(A,  lda, m0, kn, nbase,          w, lane);
            async_copy_tile(Bt, ldb, n0, kn, nbase + 10240u, w, lane);
            wait_async4();   // in-order completion => tile kt resident
        } else {
            wait_async0();
        }
        __syncthreads();
        const _Float16* cA = smem + (kt & 1) * (2 * TSZ);
        const _Float16* cB = cA + TSZ;
        mma_step(cA, 40, 0, cB, 40, 0, wm, wn, lane, acc);
        __syncthreads();     // all waves done reading before buffer reuse
    }

    const int half = lane >> 4, ml = lane & 15;
#pragma unroll
    for (int bn = 0; bn < 4; ++bn) {
        const int col = n0 + wn * 64 + bn * 16 + ml;
        const float bv = bias[col];
#pragma unroll
        for (int am = 0; am < 2; ++am) {
            const int rbase = m0 + wm * 32 + am * 16 + half * 8;
#pragma unroll
            for (int r = 0; r < 8; ++r) {
                const float val = acc[am][bn][r] + bv;
                if (OUTF32)
                    outf[(size_t)(rbase + r) * ldo + col] = val;
                else
                    outh[(size_t)(rbase + r) * ldo + col] = (_Float16)val;
            }
        }
    }
}

// ---------------------------------------------------------------------------
// KV partials: per (b,h,split) compute 128x128 block of sum_s K[s,d]*V[s,e]
// over a T/8 slice. Operands transposed on LDS write so both become [feat][s].
// ---------------------------------------------------------------------------
__global__ __launch_bounds__(256) void kv_partial(const _Float16* __restrict__ K16,
                                                  const _Float16* __restrict__ V16,
                                                  float* __restrict__ part) {
    __shared__ _Float16 sKt[128 * 72];
    __shared__ _Float16 sVt[128 * 72];
    const int tid = threadIdx.x, lane = tid & 31, w = tid >> 5;
    const int wm = w & 3, wn = w >> 2;
    const int bh = blockIdx.x, split = blockIdx.y;
    const int b = bh >> 4, h = bh & 15;
    const size_t head = (size_t)b * TT * EE + (size_t)h * DD;
    const int sBeg = split * (TT / SPLIT);

    v8f acc[2][4] = {};

    for (int s0 = 0; s0 < TT / SPLIT; s0 += 64) {
        const int d0 = (tid & 15) * 8;  // 16 threads per s-row, 8 contiguous d each
#pragma unroll
        for (int p = 0; p < 4; ++p) {
            const int sr = (tid >> 4) + p * 16;
            const size_t g = head + (size_t)(sBeg + s0 + sr) * EE + d0;
            v8h kv = *(const v8h*)&K16[g];
            v8h vv = *(const v8h*)&V16[g];
#pragma unroll
            for (int j = 0; j < 8; ++j) {
                sKt[(d0 + j) * 72 + sr] = kv[j];
                sVt[(d0 + j) * 72 + sr] = vv[j];
            }
        }
        __syncthreads();
        mma_step(sKt, 72, 0,  sVt, 72, 0,  wm, wn, lane, acc);
        mma_step(sKt, 72, 32, sVt, 72, 32, wm, wn, lane, acc);
        __syncthreads();
    }

    const int half = lane >> 4, ml = lane & 15;
    float* dst = part + ((size_t)split * 64 + bh) * (DD * DD);
#pragma unroll
    for (int am = 0; am < 2; ++am)
#pragma unroll
        for (int bn = 0; bn < 4; ++bn)
#pragma unroll
            for (int r = 0; r < 8; ++r) {
                const int drow = wm * 32 + am * 16 + half * 8 + r;
                const int ecol = wn * 64 + bn * 16 + ml;
                dst[(size_t)drow * DD + ecol] = acc[am][bn][r];
            }
}

// Sum the 8 T-split partials, fold scale=1/sqrt(D), store KV^T as [bh][e][d] f16.
__global__ void kv_reduce(const float* __restrict__ part, _Float16* __restrict__ kvt) {
    const int idx = blockIdx.x * 256 + threadIdx.x;   // 64*128*128 total
    const int bh = idx >> 14;
    const int rem = idx & 16383;                      // d*128 + e
    const int d = rem >> 7, e = rem & 127;
    float s = 0.f;
#pragma unroll
    for (int sp = 0; sp < SPLIT; ++sp)
        s += part[((size_t)sp * 64 + bh) * 16384 + rem];
    const float scale = 0.08838834764831845f;         // 1/sqrt(128)
    kvt[((size_t)bh << 14) + (size_t)e * DD + d] = (_Float16)(s * scale);
}

// ---------------------------------------------------------------------------
// attn[t, h*D+e] = sum_d Q[t, h*D+d] * KV[d,e]   (KV^T resident in LDS)
// Grid: (T/128, B*H). Block computes 128 tokens x 128 outputs of one head.
// ---------------------------------------------------------------------------
__global__ __launch_bounds__(256) void attn_apply(const _Float16* __restrict__ Q16,
                                                  const _Float16* __restrict__ kvt,
                                                  _Float16* __restrict__ attn) {
    __shared__ _Float16 sQ[128 * 72];
    __shared__ _Float16 sKV[128 * 136];
    const int tid = threadIdx.x, lane = tid & 31, w = tid >> 5;
    const int wm = w & 3, wn = w >> 2;
    const int bh = blockIdx.y, b = bh >> 4, h = bh & 15;
    const int t0 = blockIdx.x * 128;
    const _Float16* Qh = Q16 + (size_t)b * TT * EE + (size_t)h * DD;
    const _Float16* KVh = kvt + ((size_t)bh << 14);   // [e][d], ld = 128

    {   // load full 128x128 KV^T
        const int r = tid >> 3, c = (tid & 7) * 16;
#pragma unroll
        for (int p = 0; p < 4; ++p)
            *(v16h*)&sKV[(r + p * 32) * 136 + c] =
                *(const v16h*)&KVh[(size_t)(r + p * 32) * DD + c];
    }

    v8f acc[2][4] = {};
    for (int k0 = 0; k0 < DD; k0 += 64) {
        const int r = tid >> 2, c = (tid & 3) * 16;
#pragma unroll
        for (int p = 0; p < 2; ++p)
            *(v16h*)&sQ[(r + p * 64) * 72 + c] =
                *(const v16h*)&Qh[(size_t)(t0 + r + p * 64) * EE + k0 + c];
        __syncthreads();
        mma_step(sQ, 72, 0,  sKV, 136, k0,      wm, wn, lane, acc);
        mma_step(sQ, 72, 32, sKV, 136, k0 + 32, wm, wn, lane, acc);
        __syncthreads();
    }

    const int half = lane >> 4, ml = lane & 15;
#pragma unroll
    for (int am = 0; am < 2; ++am)
#pragma unroll
        for (int bn = 0; bn < 4; ++bn)
#pragma unroll
            for (int r = 0; r < 8; ++r) {
                const int trow = t0 + wm * 32 + am * 16 + half * 8 + r;
                const int ecol = wn * 64 + bn * 16 + ml;
                attn[((size_t)b * TT + trow) * EE + h * DD + ecol] = (_Float16)acc[am][bn][r];
            }
}

// ---------------------------------------------------------------------------
extern "C" void kernel_launch(void* const* d_in, const int* in_sizes, int n_in,
                              void* d_out, int out_size, void* d_ws, size_t ws_size,
                              hipStream_t stream) {
    const float* x  = (const float*)d_in[0];
    const float* Wq = (const float*)d_in[1];
    const float* bq = (const float*)d_in[2];
    const float* Wk = (const float*)d_in[3];
    const float* bk = (const float*)d_in[4];
    const float* Wv = (const float*)d_in[5];
    const float* bv = (const float*)d_in[6];
    const float* Wo = (const float*)d_in[7];
    const float* bo = (const float*)d_in[8];
    float* out = (float*)d_out;

    char* ws = (char*)d_ws;
    _Float16* x16    = (_Float16*)(ws + O_X16);
    _Float16* attn16 = (_Float16*)(ws + O_X16);   // alias: x16 dead after QKV GEMMs
    _Float16* wq16   = (_Float16*)(ws + O_WQ16);
    _Float16* wk16   = (_Float16*)(ws + O_WK16);
    _Float16* wv16   = (_Float16*)(ws + O_WV16);
    _Float16* wo16   = (_Float16*)(ws + O_WO16);
    _Float16* q16    = (_Float16*)(ws + O_Q16);
    _Float16* k16    = (_Float16*)(ws + O_K16);
    _Float16* v16    = (_Float16*)(ws + O_V16);
    float*    kvp    = (float*)   (ws + O_KVP);
    _Float16* kvt16  = (_Float16*)(ws + O_KVT);

    const long nX = (long)BT * EE;      // 33,554,432
    const long nW = (long)EE * EE;      //  4,194,304
    cvt_f32_f16<<<(int)(nX / 1024), 256, 0, stream>>>(x,  x16,  nX);
    cvt_f32_f16<<<(int)(nW / 1024), 256, 0, stream>>>(Wq, wq16, nW);
    cvt_f32_f16<<<(int)(nW / 1024), 256, 0, stream>>>(Wk, wk16, nW);
    cvt_f32_f16<<<(int)(nW / 1024), 256, 0, stream>>>(Wv, wv16, nW);
    cvt_f32_f16<<<(int)(nW / 1024), 256, 0, stream>>>(Wo, wo16, nW);

    dim3 gg(EE / 128, BT / 128);        // 16 x 128 blocks
    gemm_nt<0><<<gg, 256, 0, stream>>>(x16, wq16, bq, q16, nullptr, EE, EE, EE, EE);
    gemm_nt<0><<<gg, 256, 0, stream>>>(x16, wk16, bk, k16, nullptr, EE, EE, EE, EE);
    gemm_nt<0><<<gg, 256, 0, stream>>>(x16, wv16, bv, v16, nullptr, EE, EE, EE, EE);

    kv_partial<<<dim3(64, SPLIT), 256, 0, stream>>>(k16, v16, kvp);
    kv_reduce<<<(64 * DD * DD) / 256, 256, 0, stream>>>(kvp, kvt16);

    attn_apply<<<dim3(TT / 128, 64), 256, 0, stream>>>(q16, kvt16, attn16);

    gemm_nt<1><<<gg, 256, 0, stream>>>(attn16, wo16, bo, nullptr, out, EE, EE, EE, EE);
}
